// GCN_36112085024795
// MI455X (gfx1250) — matrix operand order repaired and checked by reading, not compiled, not run on hardware
//
#include <hip/hip_runtime.h>
#include <hip/hip_bf16.h>
#include <stdint.h>

typedef __attribute__((ext_vector_type(16))) _Float16 v16h;
typedef __attribute__((ext_vector_type(8)))  _Float16 v8h;
typedef __attribute__((ext_vector_type(4)))  _Float16 v4h;
typedef __attribute__((ext_vector_type(8)))  float    v8f;

#define N_NODES 50000
#define N_EDGES 800000
#define NFEAT   512
#define NHID    256
#define NOUT    64

union V16U { v16h v; v8h h[2]; };

// ---------------- cast kernels ----------------

__global__ void cast_f32_to_f16(const float* __restrict__ in, _Float16* __restrict__ out, long n) {
    long i = ((long)blockIdx.x * blockDim.x + threadIdx.x) * 4;
    if (i + 3 >= n) {
        for (long j = i; j < n; ++j) out[j] = (_Float16)in[j];
        return;
    }
    float4 v = *reinterpret_cast<const float4*>(in + i);
    v4h o; o[0] = (_Float16)v.x; o[1] = (_Float16)v.y; o[2] = (_Float16)v.z; o[3] = (_Float16)v.w;
    *reinterpret_cast<v4h*>(out + i) = o;
}

// in: [K][N] row-major f32 -> out: [N][K] f16 (N-major for WMMA B operand)
__global__ void cast_transpose_f16(const float* __restrict__ in, _Float16* __restrict__ out,
                                   int K, int N) {
    int idx = blockIdx.x * blockDim.x + threadIdx.x;
    if (idx >= K * N) return;
    int k = idx / N, n = idx - k * N;
    out[(size_t)n * K + k] = (_Float16)in[idx];
}

// ---------------- CSR row pointer from sorted row[] ----------------

__global__ void build_row_ptr(const int* __restrict__ row, int* __restrict__ rp,
                              int nNodes, int nEdges) {
    int n = blockIdx.x * blockDim.x + threadIdx.x;
    if (n > nNodes) return;
    int lo = 0, hi = nEdges;
    while (lo < hi) {                 // lower_bound: first e with row[e] >= n
        int mid = (lo + hi) >> 1;
        if (row[mid] < n) lo = mid + 1; else hi = mid;
    }
    rp[n] = lo;
}

// ---------------- WMMA GEMM with async-LDS double-buffered B staging ----------------
// C[M x Nc] = A[M x K](f16,row-major) * Bt[Nc x K](f16,N-major)
// Block = 256 threads = 8 waves; each wave computes a 16(M) x 64(N) strip.
// B strip (64 N x 64 K per stage) staged into LDS with global_load_async_to_lds_b128,
// double-buffered so the async copy overlaps the WMMA work of the previous chunk.

template <int K>
__global__ __launch_bounds__(256) void gemm_wmma_f16_lds(
    const _Float16* __restrict__ A, const _Float16* __restrict__ Bt,
    float* __restrict__ C, int M, int ldc) {
    constexpr int BK   = 64;              // K per stage (2 WMMA k-steps)
    constexpr int ROWH = BK + 8;          // padded LDS row stride (halves) -> 144 B
    __shared__ _Float16 Bs[2][64 * ROWH]; // 2 x 9 KB

    const int tid  = threadIdx.x;
    const int lane = tid & 31;
    const int wave = tid >> 5;
    const int rowBase = blockIdx.x * 128 + wave * 16;
    const bool active = (rowBase < M);
    const int nBase = blockIdx.y * 64;

    // staging coords: 512 x 16B segments per stage, 2 per thread
    const int segk = (tid & 7) * 8;       // k offset (halves) of 16B segment
    const int nA   = tid >> 3;            // row 0..31 (j adds 32)
    const _Float16* gsrcBase = Bt + (size_t)nBase * K;

    auto stage = [&](int buf, int kk0) {
#pragma unroll
        for (int j = 0; j < 2; ++j) {
            const int n = nA + j * 32;
            unsigned lds = (unsigned)(uintptr_t)&Bs[buf][n * ROWH + segk];
            const _Float16* g = gsrcBase + (size_t)n * K + kk0 + segk;
            asm volatile("global_load_async_to_lds_b128 %0, %1, off"
                         :: "v"(lds), "v"(g) : "memory");
        }
    };

    const int rA     = lane & 15;          // M (and B's N) index within tile
    const int kSel8  = (lane >> 4) * 8;    // A: half-lane K sub-run
    const int kSel16 = (lane >> 4) * 16;   // unused; B is read with kSel8 like A
    const _Float16* aRow = A + (size_t)((active ? rowBase : 0) + rA) * K;

    v8f acc[4] = {};

    stage(0, 0);
    int buf = 0;
    for (int kk = 0; kk < K; kk += BK) {
        asm volatile("s_wait_asynccnt 0x0" ::: "memory"); // my stage writes landed
        __syncthreads();                                  // everyone's landed
        if (kk + BK < K) stage(buf ^ 1, kk + BK);         // prefetch next chunk

#pragma unroll
        for (int ks = 0; ks < BK; ks += 32) {
            V16U a;
            const _Float16* ap = aRow + kk + ks + kSel8;
            a.h[0] = *reinterpret_cast<const v8h*>(ap);        // K = +0..7
            a.h[1] = *reinterpret_cast<const v8h*>(ap + 16);   // K = +16..23
#pragma unroll
            for (int nt = 0; nt < 4; ++nt) {
                const _Float16* bp = &Bs[buf][(nt * 16 + rA) * ROWH + ks + kSel8];
                V16U b;
                b.h[0] = *reinterpret_cast<const v8h*>(bp);
                b.h[1] = *reinterpret_cast<const v8h*>(bp + 16);
                acc[nt] = __builtin_amdgcn_wmma_f32_16x16x32_f16(
                    false, a.v, false, b.v, (short)0, acc[nt], false, false);
            }
        }
        buf ^= 1;
    }

    if (!active) return;
    const int cCol  = nBase + rA;
    const int rbase = rowBase + ((lane >> 4) << 3);
#pragma unroll
    for (int nt = 0; nt < 4; ++nt)
#pragma unroll
        for (int r = 0; r < 8; ++r)
            C[(size_t)(rbase + r) * ldc + cCol + nt * 16] = acc[nt][r];
}

// ---------------- SpMM layer 1: ah = f16(relu(segment_sum(h[col]*ew) + b1)) ----------------
// One wave per destination node; lane owns 8 contiguous f32 features (32 B).
// node forced wave-uniform so col[e]/ew[e]/rp[] become scalar (SMEM) loads.

__global__ __launch_bounds__(256) void spmm1_relu_f16(
    const float* __restrict__ h, const float* __restrict__ ew,
    const int* __restrict__ col, const int* __restrict__ rp,
    const float* __restrict__ b1, _Float16* __restrict__ ah, int nNodes) {
    const int lane = threadIdx.x & 31;
    const int node = __builtin_amdgcn_readfirstlane(blockIdx.x * 8 + (threadIdx.x >> 5));
    if (node >= nNodes) return;
    const int e0 = rp[node], e1 = rp[node + 1];
    const int f0 = lane * 8;

    float acc[8] = {};
    int e = e0;
    for (; e + 1 < e1; e += 2) {
        const int   c0 = col[e],     c1 = col[e + 1];
        const float w0 = ew[e],      w1 = ew[e + 1];
        const float4* p0 = reinterpret_cast<const float4*>(h + (size_t)c0 * NHID + f0);
        const float4* p1 = reinterpret_cast<const float4*>(h + (size_t)c1 * NHID + f0);
        float4 a0 = p0[0], a1 = p0[1], b0 = p1[0], b1v = p1[1];
        acc[0] += w0 * a0.x; acc[1] += w0 * a0.y; acc[2] += w0 * a0.z; acc[3] += w0 * a0.w;
        acc[4] += w0 * a1.x; acc[5] += w0 * a1.y; acc[6] += w0 * a1.z; acc[7] += w0 * a1.w;
        acc[0] += w1 * b0.x; acc[1] += w1 * b0.y; acc[2] += w1 * b0.z; acc[3] += w1 * b0.w;
        acc[4] += w1 * b1v.x; acc[5] += w1 * b1v.y; acc[6] += w1 * b1v.z; acc[7] += w1 * b1v.w;
    }
    if (e < e1) {
        const int   c = col[e];
        const float w = ew[e];
        const float4* p = reinterpret_cast<const float4*>(h + (size_t)c * NHID + f0);
        float4 v0 = p[0], v1 = p[1];
        acc[0] += w * v0.x; acc[1] += w * v0.y; acc[2] += w * v0.z; acc[3] += w * v0.w;
        acc[4] += w * v1.x; acc[5] += w * v1.y; acc[6] += w * v1.z; acc[7] += w * v1.w;
    }
    v8h o;
#pragma unroll
    for (int j = 0; j < 8; ++j) {
        float t = acc[j] + b1[f0 + j];
        o[j] = (_Float16)fmaxf(t, 0.0f);
    }
    *reinterpret_cast<v8h*>(ah + (size_t)node * NHID + f0) = o;
}

// ---------------- SpMM layer 2: out = segment_sum(h2[col]*ew) + b2 (f32) ----------------

__global__ __launch_bounds__(256) void spmm2_bias(
    const float* __restrict__ h2, const float* __restrict__ ew,
    const int* __restrict__ col, const int* __restrict__ rp,
    const float* __restrict__ b2, float* __restrict__ out, int nNodes) {
    const int lane = threadIdx.x & 31;
    const int node = __builtin_amdgcn_readfirstlane(blockIdx.x * 8 + (threadIdx.x >> 5));
    if (node >= nNodes) return;
    const int e0 = rp[node], e1 = rp[node + 1];
    const int f0 = lane * 2;

    float a0 = 0.f, a1 = 0.f;
    int e = e0;
    for (; e + 1 < e1; e += 2) {
        const int   c0 = col[e],  c1 = col[e + 1];
        const float w0 = ew[e],   w1 = ew[e + 1];
        const float2 v0 = *reinterpret_cast<const float2*>(h2 + (size_t)c0 * NOUT + f0);
        const float2 v1 = *reinterpret_cast<const float2*>(h2 + (size_t)c1 * NOUT + f0);
        a0 += w0 * v0.x + w1 * v1.x;
        a1 += w0 * v0.y + w1 * v1.y;
    }
    if (e < e1) {
        const int   c = col[e];
        const float w = ew[e];
        const float2 v = *reinterpret_cast<const float2*>(h2 + (size_t)c * NOUT + f0);
        a0 += w * v.x; a1 += w * v.y;
    }
    float2 r; r.x = a0 + b2[f0]; r.y = a1 + b2[f0 + 1];
    *reinterpret_cast<float2*>(out + (size_t)node * NOUT + f0) = r;
}

// ---------------- host launcher ----------------

extern "C" void kernel_launch(void* const* d_in, const int* in_sizes, int n_in,
                              void* d_out, int out_size, void* d_ws, size_t ws_size,
                              hipStream_t stream) {
    const float* x    = (const float*)d_in[0];
    const float* w1   = (const float*)d_in[1];
    const float* b1   = (const float*)d_in[2];
    const float* w2   = (const float*)d_in[3];
    const float* b2   = (const float*)d_in[4];
    const float* ew   = (const float*)d_in[5];
    const int*   row  = (const int*)d_in[6];
    const int*   col  = (const int*)d_in[7];
    float*       out  = (float*)d_out;

    // carve workspace (256B aligned chunks)
    char* p = (char*)d_ws;
    auto carve = [&](size_t bytes) -> char* {
        char* r = p;
        p += (bytes + 255) & ~(size_t)255;
        return r;
    };
    _Float16* xh  = (_Float16*)carve((size_t)N_NODES * NFEAT * 2);   // 51.2 MB
    _Float16* w1t = (_Float16*)carve((size_t)NHID * NFEAT * 2);      // 256 KB  [NHID][NFEAT]
    _Float16* w2t = (_Float16*)carve((size_t)NOUT * NHID * 2);       // 32 KB   [NOUT][NHID]
    int*      rp  = (int*)carve((size_t)(N_NODES + 1) * 4);          // 200 KB
    float*    h1  = (float*)carve((size_t)N_NODES * NHID * 4);       // 51.2 MB
    _Float16* ah  = (_Float16*)carve((size_t)N_NODES * NHID * 2);    // 25.6 MB
    float*    h2  = (float*)carve((size_t)N_NODES * NOUT * 4);       // 12.8 MB

    // 1) operand casts
    {
        long n = (long)N_NODES * NFEAT;
        int blocks = (int)((n + 1023) / 1024);
        cast_f32_to_f16<<<blocks, 256, 0, stream>>>(x, xh, n);
    }
    cast_transpose_f16<<<(NFEAT * NHID + 255) / 256, 256, 0, stream>>>(w1, w1t, NFEAT, NHID);
    cast_transpose_f16<<<(NHID * NOUT + 255) / 256, 256, 0, stream>>>(w2, w2t, NHID, NOUT);

    // 2) CSR row pointers from sorted row[]
    build_row_ptr<<<(N_NODES + 1 + 255) / 256, 256, 0, stream>>>(row, rp, N_NODES, N_EDGES);

    // 3) h1 = x @ w1   (WMMA f16 -> f32 accum), M=50000, K=512, N=256
    {
        dim3 grid((N_NODES + 127) / 128, NHID / 64);
        gemm_wmma_f16_lds<NFEAT><<<grid, 256, 0, stream>>>(xh, w1t, h1, N_NODES, NHID);
    }

    // 4) ah = f16(relu(spmm(h1) + b1))
    spmm1_relu_f16<<<(N_NODES + 7) / 8, 256, 0, stream>>>(h1, ew, col, rp, b1, ah, N_NODES);

    // 5) h2 = ah @ w2  (WMMA), M=50000, K=256, N=64
    {
        dim3 grid((N_NODES + 127) / 128, NOUT / 64);
        gemm_wmma_f16_lds<NHID><<<grid, 256, 0, stream>>>(ah, w2t, h2, N_NODES, NOUT);
    }

    // 6) out = spmm(h2) + b2
    spmm2_bias<<<(N_NODES + 7) / 8, 256, 0, stream>>>(h2, ew, col, rp, b2, out, N_NODES);
}
